// MyCausalSelfAttention_23914377904416
// MI455X (gfx1250) — matrix operand order repaired and checked by reading, not compiled, not run on hardware
//
#include <hip/hip_runtime.h>

// ---------------------------------------------------------------------------
// CDNA5 (gfx1250) bf16-WMMA causal self-attention, software-pipelined:
//   stage 0: fp32 -> bf16 conversion of x, w_attn, w_proj
//   stage 1: qkv = x @ w_attn + b_attn          (bf16 WMMA, f32 accum)
//   stage 2: flash-attention (bf16 WMMA QK^T / PV, f32 online softmax)
//   stage 3: out = y @ w_proj + b_proj          (bf16 WMMA, f32 out)
// Non-transposed tiles are staged with global_load_async_to_lds_b128
// (ASYNCcnt path); transposed tiles go global->VGPR->LDS with batched loads.
// LDS is double-buffered so global traffic for tile t+1 overlaps the WMMA
// block of tile t.
// ---------------------------------------------------------------------------

typedef __bf16 v16bf __attribute__((ext_vector_type(16)));
typedef float  v8f   __attribute__((ext_vector_type(8)));
typedef int    v4i_vs __attribute__((vector_size(16)));   // matches builtin param

union Frag16 { uint4 q[2]; v16bf v; };

#if defined(__has_builtin)
#if __has_builtin(__builtin_amdgcn_global_load_async_to_lds_b128)
#define HAVE_ASYNC_LDS 1
#endif
#endif
#ifndef HAVE_ASYNC_LDS
#define HAVE_ASYNC_LDS 0
#endif

__device__ __forceinline__ void async_copy16(const unsigned short* g, unsigned short* l) {
#if HAVE_ASYNC_LDS
  __builtin_amdgcn_global_load_async_to_lds_b128(
      (__attribute__((address_space(1))) v4i_vs*)g,
      (__attribute__((address_space(3))) v4i_vs*)l, 0, 0);
#else
  *reinterpret_cast<uint4*>(l) = *reinterpret_cast<const uint4*>(g);
#endif
}

__device__ __forceinline__ void async_wait0() {
#if HAVE_ASYNC_LDS
  asm volatile("s_wait_asynccnt 0x0" ::: "memory");
#endif
}

__device__ __forceinline__ unsigned short f32_to_bf16(float f) {
  union { float f; unsigned u; } v; v.f = f;
  unsigned r = v.u + 0x7FFFu + ((v.u >> 16) & 1u);   // round-to-nearest-even
  return (unsigned short)(r >> 16);
}

__device__ __forceinline__ v8f wmma_bf16(v16bf a, v16bf b, v8f c) {
  return __builtin_amdgcn_wmma_f32_16x16x32_bf16(false, a, false, b, (short)0, c,
                                                 false, false);
}

// ---------------------------------------------------------------------------
// fp32 -> bf16 converter (4 elements / thread, n divisible by 4)
// ---------------------------------------------------------------------------
__global__ __launch_bounds__(256) void cvt_f32_bf16(const float* __restrict__ in,
                                                    unsigned short* __restrict__ out,
                                                    int n) {
  int i = (blockIdx.x * 256 + threadIdx.x) * 4;
  if (i >= n) return;
  float4 f = *reinterpret_cast<const float4*>(in + i);
  union { unsigned short s[4]; uint2 u; } o;
  o.s[0] = f32_to_bf16(f.x);
  o.s[1] = f32_to_bf16(f.y);
  o.s[2] = f32_to_bf16(f.z);
  o.s[3] = f32_to_bf16(f.w);
  *reinterpret_cast<uint2*>(out + i) = o.u;
}

// ---------------------------------------------------------------------------
// Tiled bf16 GEMM: C[M,N] = A[M,K] * B[K,N] + bias[N]
// Block tile 128x128x32, 8 waves (4x2), wave tile 32x64 (2x4 WMMA tiles).
// Double-buffered LDS; A tile via async-to-LDS, B tile via regs (transposed).
// ---------------------------------------------------------------------------
#define GEMM_BM 128
#define GEMM_BN 128
#define GEMM_BK 32
#define GEMM_AST 40   // 80B row stride -> 20-dword bank stride, conflict-free
#define GEMM_BST 40

template <bool WF32>
__global__ __launch_bounds__(256) void gemm_bf16(
    const unsigned short* __restrict__ A,   // [M,K] bf16 row-major
    const unsigned short* __restrict__ B,   // [K,N] bf16 row-major
    const float* __restrict__ bias,         // [N]
    float* __restrict__ Cf,                 // f32 out (WF32)
    unsigned short* __restrict__ Cb,        // bf16 out (!WF32)
    int M, int N, int K) {
  __shared__ unsigned short As[2][GEMM_BM * GEMM_AST];
  __shared__ unsigned short Bs[2][GEMM_BN * GEMM_BST];

  const int tid  = threadIdx.x;
  const int lane = tid & 31;
  const int wave = tid >> 5;
  const int wm   = wave >> 1;           // 0..3 (rows)
  const int wn   = wave & 1;            // 0..1 (cols)
  const int lr   = lane & 15;
  const int lh   = lane >> 4;
  const int rowBase = blockIdx.y * GEMM_BM;
  const int colBase = blockIdx.x * GEMM_BN;
  const int nk = K / GEMM_BK;

  v8f acc[2][4];
#pragma unroll
  for (int i = 0; i < 2; ++i)
#pragma unroll
    for (int j = 0; j < 4; ++j)
      acc[i][j] = (v8f){0.f, 0.f, 0.f, 0.f, 0.f, 0.f, 0.f, 0.f};

  uint4 breg[2];

  auto issueA = [&](int kt, int buf) {   // 128x32 tile, 512 x 16B chunks
    int k0 = kt * GEMM_BK;
#pragma unroll
    for (int i = 0; i < 2; ++i) {
      int c = tid + i * 256;
      int r = c >> 2, c8 = (c & 3) << 3;
      async_copy16(A + (size_t)(rowBase + r) * K + k0 + c8,
                   &As[buf][r * GEMM_AST + c8]);
    }
  };
  auto loadB = [&](int kt) {             // 32x128 tile into regs
    int k0 = kt * GEMM_BK;
#pragma unroll
    for (int i = 0; i < 2; ++i) {
      int c = tid + i * 256;
      int kr = c >> 4, n8 = (c & 15) << 3;
      breg[i] = *reinterpret_cast<const uint4*>(B + (size_t)(k0 + kr) * N + colBase + n8);
    }
  };
  auto storeB = [&](int buf) {           // scatter transposed: Bs[n][k]
#pragma unroll
    for (int i = 0; i < 2; ++i) {
      int c = tid + i * 256;
      int kr = c >> 4, n8 = (c & 15) << 3;
      union { uint4 q; unsigned short s[8]; } u; u.q = breg[i];
#pragma unroll
      for (int j = 0; j < 8; ++j) Bs[buf][(n8 + j) * GEMM_BST + kr] = u.s[j];
    }
  };

  // ---- prologue: tile 0 into buffer 0
  issueA(0, 0);
  loadB(0);
  async_wait0();
  storeB(0);
  __syncthreads();

  for (int kt = 0; kt < nk; ++kt) {
    const int cur = kt & 1;
    if (kt + 1 < nk) {          // issue next tile's global traffic first
      issueA(kt + 1, cur ^ 1);
      loadB(kt + 1);
      if (kt + 2 < nk)          // warm L2 for the tile after that
        __builtin_prefetch(A + (size_t)(rowBase + (tid >> 1)) * K + (kt + 2) * GEMM_BK, 0, 1);
    }

    // ---- fragments from LDS buffer `cur`
    Frag16 af[2], bfr[4];
#pragma unroll
    for (int i = 0; i < 2; ++i) {
      int m = wm * 32 + i * 16 + lr;
      af[i].q[0] = *reinterpret_cast<const uint4*>(&As[cur][m * GEMM_AST + 16 * lh]);
      af[i].q[1] = *reinterpret_cast<const uint4*>(&As[cur][m * GEMM_AST + 16 * lh + 8]);
    }
#pragma unroll
    for (int j = 0; j < 4; ++j) {
      int n = wn * 64 + j * 16 + lr;
      bfr[j].q[0] = *reinterpret_cast<const uint4*>(&Bs[cur][n * GEMM_BST + 16 * lh]);
      bfr[j].q[1] = *reinterpret_cast<const uint4*>(&Bs[cur][n * GEMM_BST + 16 * lh + 8]);
    }
#pragma unroll
    for (int i = 0; i < 2; ++i)
#pragma unroll
      for (int j = 0; j < 4; ++j)
        acc[i][j] = wmma_bf16(af[i].v, bfr[j].v, acc[i][j]);

    if (kt + 1 < nk) {
      async_wait0();            // next A tile landed in LDS
      storeB(cur ^ 1);          // next B tile into the other buffer
      __syncthreads();
    }
  }

  // ---- epilogue: C layout element r -> (m = 8*lh + r, n = lr)
#pragma unroll
  for (int i = 0; i < 2; ++i) {
#pragma unroll
    for (int j = 0; j < 4; ++j) {
      int n = colBase + wn * 64 + j * 16 + lr;
      float bv = bias[n];
#pragma unroll
      for (int r = 0; r < 8; ++r) {
        int m = rowBase + wm * 32 + i * 16 + lh * 8 + r;
        float val = acc[i][j][r] + bv;
        size_t idx = (size_t)m * N + n;
        if (WF32) Cf[idx] = val;
        else      Cb[idx] = f32_to_bf16(val);
      }
    }
  }
}

// ---------------------------------------------------------------------------
// Flash attention: one workgroup = one (b,h) x 128 query rows, 8 waves.
// K tile async-to-LDS row-major, V tile via regs transposed; both
// double-buffered across the key loop. P round-trips through a per-wave LDS
// slab to reach A-fragment layout for the PV WMMAs.
// ---------------------------------------------------------------------------
#define ATT_T  2048
#define ATT_C  1024
#define ATT_C3 3072
#define ATT_D  64
#define ATT_KST 72   // 144B row stride -> 36-dword bank stride, conflict-free

__global__ __launch_bounds__(256) void attn_fa(
    const unsigned short* __restrict__ qkv,   // [B,T,3C] bf16
    unsigned short* __restrict__ y) {         // [B,T,C]  bf16
  __shared__ unsigned short Ks[2][64 * ATT_KST];   // K tile  [key][d]
  __shared__ unsigned short Vs[2][64 * ATT_KST];   // V tile  [d][key]
  __shared__ unsigned short Ps[8 * 16 * ATT_KST];  // per-wave P slab [16][64]

  const int tid  = threadIdx.x;
  const int lane = tid & 31;
  const int wave = tid >> 5;
  const int lr   = lane & 15;
  const int lh   = lane >> 4;
  const int b    = blockIdx.y >> 4;   // H = 16
  const int h    = blockIdx.y & 15;
  const int qBase = blockIdx.x * 128;
  const int qRow0 = qBase + wave * 16;

  // ---- Q fragments (A layout), kept in registers for the whole pass
  Frag16 qf[2];
  {
    int t = qRow0 + lr;
    const unsigned short* qp = qkv + (size_t)(b * ATT_T + t) * ATT_C3 + h * ATT_D;
#pragma unroll
    for (int hf = 0; hf < 2; ++hf) {
      qf[hf].q[0] = *reinterpret_cast<const uint4*>(qp + hf * 32 + 16 * lh);
      qf[hf].q[1] = *reinterpret_cast<const uint4*>(qp + hf * 32 + 16 * lh + 8);
    }
  }

  v8f acc[4];
#pragma unroll
  for (int dt = 0; dt < 4; ++dt)
    acc[dt] = (v8f){0.f, 0.f, 0.f, 0.f, 0.f, 0.f, 0.f, 0.f};
  float mrow[8], lrow[8];
#pragma unroll
  for (int r = 0; r < 8; ++r) { mrow[r] = -1e30f; lrow[r] = 0.0f; }

  const unsigned short* kbase = qkv + (size_t)b * ATT_T * ATT_C3 + ATT_C + h * ATT_D;
  const unsigned short* vbase = qkv + (size_t)b * ATT_T * ATT_C3 + 2 * ATT_C + h * ATT_D;
  const int nkt = (qBase >> 6) + 2;   // causal: keys only up to qBase+127

  uint4 vreg[2];
  auto issueK = [&](int kt, int buf) {
#pragma unroll
    for (int i = 0; i < 2; ++i) {
      int c = tid + i * 256;
      int key = c >> 3, d8 = (c & 7) << 3;
      size_t grow = (size_t)(kt * 64 + key) * ATT_C3;
      async_copy16(kbase + grow + d8, &Ks[buf][key * ATT_KST + d8]);
    }
  };
  auto loadV = [&](int kt) {
#pragma unroll
    for (int i = 0; i < 2; ++i) {
      int c = tid + i * 256;
      int key = c >> 3, d8 = (c & 7) << 3;
      size_t grow = (size_t)(kt * 64 + key) * ATT_C3;
      vreg[i] = *reinterpret_cast<const uint4*>(vbase + grow + d8);
    }
  };
  auto storeV = [&](int buf) {
#pragma unroll
    for (int i = 0; i < 2; ++i) {
      int c = tid + i * 256;
      int key = c >> 3, d8 = (c & 7) << 3;
      union { uint4 q; unsigned short s[8]; } u; u.q = vreg[i];
#pragma unroll
      for (int j = 0; j < 8; ++j) Vs[buf][(d8 + j) * ATT_KST + key] = u.s[j];
    }
  };

  // ---- prologue: key tile 0 into buffer 0
  issueK(0, 0);
  loadV(0);
  async_wait0();
  storeV(0);
  __syncthreads();

  for (int kt = 0; kt < nkt; ++kt) {
    const int cur = kt & 1;
    if (kt + 1 < nkt) {        // overlap next tile's global traffic with compute
      issueK(kt + 1, cur ^ 1);
      loadV(kt + 1);
    }

    // ---- S = Q K^T for 4 key sub-tiles of 16
    v8f s[4];
#pragma unroll
    for (int nt = 0; nt < 4; ++nt) {
      int keyrow = nt * 16 + lr;
      Frag16 kf0, kf1;
      kf0.q[0] = *reinterpret_cast<const uint4*>(&Ks[cur][keyrow * ATT_KST + 16 * lh]);
      kf0.q[1] = *reinterpret_cast<const uint4*>(&Ks[cur][keyrow * ATT_KST + 16 * lh + 8]);
      kf1.q[0] = *reinterpret_cast<const uint4*>(&Ks[cur][keyrow * ATT_KST + 32 + 16 * lh]);
      kf1.q[1] = *reinterpret_cast<const uint4*>(&Ks[cur][keyrow * ATT_KST + 32 + 16 * lh + 8]);
      v8f sv = (v8f){0.f, 0.f, 0.f, 0.f, 0.f, 0.f, 0.f, 0.f};
      sv = wmma_bf16(qf[0].v, kf0.v, sv);
      sv = wmma_bf16(qf[1].v, kf1.v, sv);
      s[nt] = sv;
    }

    // ---- mask + scale + online softmax (reduce over 16-lane N groups)
#pragma unroll
    for (int r = 0; r < 8; ++r) {
      int qrow = qRow0 + lh * 8 + r;
      float mx = -1e30f;
#pragma unroll
      for (int nt = 0; nt < 4; ++nt) {
        int key = kt * 64 + nt * 16 + lr;
        float v = (key <= qrow) ? s[nt][r] * 0.125f : -1e30f;   // 1/sqrt(64)
        s[nt][r] = v;
        mx = fmaxf(mx, v);
      }
#pragma unroll
      for (int off = 1; off < 16; off <<= 1)
        mx = fmaxf(mx, __shfl_xor(mx, off, 32));
      float nm = fmaxf(mrow[r], mx);
      float alpha = __expf(mrow[r] - nm);
      float rs = 0.0f;
#pragma unroll
      for (int nt = 0; nt < 4; ++nt) {
        float p = __expf(s[nt][r] - nm);
        s[nt][r] = p;
        rs += p;
      }
#pragma unroll
      for (int off = 1; off < 16; off <<= 1)
        rs += __shfl_xor(rs, off, 32);
      lrow[r] = lrow[r] * alpha + rs;
      mrow[r] = nm;
#pragma unroll
      for (int dt = 0; dt < 4; ++dt) acc[dt][r] *= alpha;
    }

    // ---- P (C layout) -> per-wave LDS slab -> A fragments
    unsigned short* pw = &Ps[wave * 16 * ATT_KST];
#pragma unroll
    for (int nt = 0; nt < 4; ++nt)
#pragma unroll
      for (int r = 0; r < 8; ++r)
        pw[(lh * 8 + r) * ATT_KST + nt * 16 + lr] = f32_to_bf16(s[nt][r]);

    Frag16 pf[2];
#pragma unroll
    for (int hf = 0; hf < 2; ++hf) {
      pf[hf].q[0] = *reinterpret_cast<const uint4*>(&pw[lr * ATT_KST + hf * 32 + 16 * lh]);
      pf[hf].q[1] = *reinterpret_cast<const uint4*>(&pw[lr * ATT_KST + hf * 32 + 16 * lh + 8]);
    }

    // ---- y += P V  (V transposed in LDS -> contiguous B fragments)
#pragma unroll
    for (int dt = 0; dt < 4; ++dt) {
      int d = dt * 16 + lr;
      Frag16 vf0, vf1;
      vf0.q[0] = *reinterpret_cast<const uint4*>(&Vs[cur][d * ATT_KST + 16 * lh]);
      vf0.q[1] = *reinterpret_cast<const uint4*>(&Vs[cur][d * ATT_KST + 16 * lh + 8]);
      vf1.q[0] = *reinterpret_cast<const uint4*>(&Vs[cur][d * ATT_KST + 32 + 16 * lh]);
      vf1.q[1] = *reinterpret_cast<const uint4*>(&Vs[cur][d * ATT_KST + 32 + 16 * lh + 8]);
      acc[dt] = wmma_bf16(pf[0].v, vf0.v, acc[dt]);
      acc[dt] = wmma_bf16(pf[1].v, vf1.v, acc[dt]);
    }

    if (kt + 1 < nkt) {
      async_wait0();            // next K tile landed
      storeV(cur ^ 1);          // next V tile (transposed) into other buffer
      __syncthreads();
    }
  }

  // ---- normalize and store y (bf16), layout [B,T,C] with head offset
#pragma unroll
  for (int r = 0; r < 8; ++r) {
    float inv = 1.0f / lrow[r];
    int t = qRow0 + lh * 8 + r;
    unsigned short* yp = y + (size_t)(b * ATT_T + t) * ATT_C + h * ATT_D;
#pragma unroll
    for (int dt = 0; dt < 4; ++dt)
      yp[dt * 16 + lr] = f32_to_bf16(acc[dt][r] * inv);
  }
}

// ---------------------------------------------------------------------------
// Launcher
// ---------------------------------------------------------------------------
extern "C" void kernel_launch(void* const* d_in, const int* in_sizes, int n_in,
                              void* d_out, int out_size, void* d_ws, size_t ws_size,
                              hipStream_t stream) {
  (void)in_sizes; (void)n_in; (void)out_size; (void)ws_size;
  constexpr int Bz = 4, T = 2048, C = 1024, H = 16;
  constexpr int M = Bz * T;         // 8192
  constexpr int C3 = 3 * C;         // 3072

  const float* x      = (const float*)d_in[0];
  const float* w_attn = (const float*)d_in[1];
  const float* b_attn = (const float*)d_in[2];
  const float* w_proj = (const float*)d_in[3];
  const float* b_proj = (const float*)d_in[4];
  float* out = (float*)d_out;

  // workspace carve (bytes, 16B-aligned)
  char* ws = (char*)d_ws;
  unsigned short* xb   = (unsigned short*)(ws);                     // 16 MiB
  unsigned short* wab  = (unsigned short*)(ws + 16777216);          //  6 MiB
  unsigned short* wpb  = (unsigned short*)(ws + 23068672);          //  2 MiB
  unsigned short* qkvb = (unsigned short*)(ws + 25165824);          // 48 MiB
  unsigned short* yb   = (unsigned short*)(ws + 75497472);          // 16 MiB

  // stage 0: fp32 -> bf16
  cvt_f32_bf16<<<(M * C) / 1024, 256, 0, stream>>>(x, xb, M * C);
  cvt_f32_bf16<<<(C * C3) / 1024, 256, 0, stream>>>(w_attn, wab, C * C3);
  cvt_f32_bf16<<<(C * C) / 1024, 256, 0, stream>>>(w_proj, wpb, C * C);

  // stage 1: qkv = x @ w_attn + b_attn   (bf16 out for attention)
  gemm_bf16<false><<<dim3(C3 / 128, M / 128), 256, 0, stream>>>(
      xb, wab, b_attn, nullptr, qkvb, M, C3, C);

  // stage 2: flash attention -> y (bf16)
  attn_fa<<<dim3(T / 128, Bz * H), 256, 0, stream>>>(qkvb, yb);

  // stage 3: out = y @ w_proj + b_proj   (f32 out)
  gemm_bf16<true><<<dim3(C / 128, M / 128), 256, 0, stream>>>(
      yb, wpb, b_proj, out, nullptr, M, C, C);
}
